// Attention_msa_4664334483722
// MI455X (gfx1250) — compile-verified, matrix operand order checked by reading, not run on hardware
//
#include <hip/hip_runtime.h>

// CDNA5 / gfx1250: wave32, bf16 WMMA 16x16x32 with f32 accumulate.
typedef __attribute__((ext_vector_type(16))) __bf16 v16bf;
typedef __attribute__((ext_vector_type(8)))  float  v8f;

#define N_TOK 2000   // 125 * 16 : exact 16-row/col tiling
#define N_PAD 2048
#define CDIM  256
#define NH    8
#define HD    32     // == K of v_wmma_f32_16x16x32_bf16 : one WMMA per score tile

// ---- WMMA fragment element -> K mapping (ISA 7.12.2, 16-bit A 16x32) ----
// lanes 0-15: VGPR0..3 hold K={0..7}, VGPR4..7 hold K={16..23}; lanes 16-31: +8
__device__ __forceinline__ int kmap(int e, int half) {
  int vg = e >> 1, lo = e & 1;
  return ((vg < 4) ? (vg * 2) : (16 + (vg - 4) * 2)) + 8 * half + lo;
}

__device__ __forceinline__ v8f wmma_bf16(v16bf a, v16bf b, v8f c) {
  // (neg_a, A, neg_b, B, c_mod, C, reuse_a, reuse_b)
  return __builtin_amdgcn_wmma_f32_16x16x32_bf16(false, a, false, b, (short)0, c,
                                                 false, false);
}

// A fragment (M=rows, K=cols) from row-major bf16 src. Also used for B = srcT
// (q.kT style: B[k][n] = src[n][k]) since the roles of lane/element swap.
__device__ __forceinline__ v16bf loadA_bf16(const __bf16* src, int stride,
                                            int r0, int k0, int lane) {
  int r = r0 + (lane & 15), half = lane >> 4;
  v16bf f;
#pragma unroll
  for (int e = 0; e < 16; ++e) f[e] = src[(size_t)r * stride + k0 + kmap(e, half)];
  return f;
}

// A fragment from a row-major f32 source (converted to bf16 on the fly)
__device__ __forceinline__ v16bf loadA_f32(const float* src, int stride,
                                           int r0, int k0, int lane) {
  int r = r0 + (lane & 15), half = lane >> 4;
  v16bf f;
#pragma unroll
  for (int e = 0; e < 16; ++e)
    f[e] = (__bf16)src[(size_t)r * stride + k0 + kmap(e, half)];
  return f;
}

// B fragment (K=rows of src, N=cols of src) from row-major f32 src
__device__ __forceinline__ v16bf loadB_f32(const float* src, int stride,
                                           int k0, int n0, int lane) {
  int n = n0 + (lane & 15), half = lane >> 4;
  v16bf f;
#pragma unroll
  for (int e = 0; e < 16; ++e)
    f[e] = (__bf16)src[(size_t)(k0 + kmap(e, half)) * stride + n];
  return f;
}

// B fragment from row-major bf16 src with row clamp (tail slab padding; the
// matching A values are zero there so the clamped rows contribute nothing)
__device__ __forceinline__ v16bf loadB_bf16_clamp(const __bf16* src, int stride,
                                                  int k0, int n0, int lane) {
  int n = n0 + (lane & 15), half = lane >> 4;
  v16bf f;
#pragma unroll
  for (int e = 0; e < 16; ++e) {
    int k = k0 + kmap(e, half);
    k = (k < N_TOK) ? k : (N_TOK - 1);
    f[e] = src[(size_t)k * stride + n];
  }
  return f;
}

// ---------------- Kernel 1: QKV GEMM  [2000x256] @ [256x768] (x2) ----------
__global__ __launch_bounds__(256) void k_qkv_gemm(
    const float* __restrict__ x_cls, const float* __restrict__ x_reg,
    const float* __restrict__ w_cls, const float* __restrict__ w_reg,
    float* __restrict__ qkv_cls, float* __restrict__ qkv_reg) {
  const int lane = threadIdx.x & 31;
  const int wid  = (blockIdx.x * blockDim.x + threadIdx.x) >> 5;
  const int tiles = 125 * 48;                 // 16x16 tiles per matrix
  if (wid >= 2 * tiles) return;
  const int mat = wid / tiles, t = wid % tiles;
  const int r0 = (t / 48) * 16, c0 = (t % 48) * 16;
  const float* X = mat ? x_reg : x_cls;
  const float* W = mat ? w_reg : w_cls;
  float* O = mat ? qkv_reg : qkv_cls;

  v8f acc = {};
#pragma unroll
  for (int kk = 0; kk < CDIM; kk += 32) {
    v16bf a = loadA_f32(X, CDIM, r0, kk, lane);
    v16bf b = loadB_f32(W, 3 * CDIM, kk, c0, lane);
    acc = wmma_bf16(a, b, acc);
  }
  const int n = c0 + (lane & 15), half = lane >> 4;
#pragma unroll
  for (int i = 0; i < 8; ++i)
    O[(size_t)(r0 + i + 8 * half) * (3 * CDIM) + n] = acc[i];
}

// ------------- Kernel 1b: per-(n,h) L2 norm + bf16 pack (+ x_ori) ----------
__global__ __launch_bounds__(256) void k_norm_pack(
    const float* __restrict__ qkv, int is_cls,
    __bf16* __restrict__ qn, __bf16* __restrict__ kn,
    __bf16* __restrict__ vn, __bf16* __restrict__ vbf,
    float* __restrict__ out_x) {
  const int lane = threadIdx.x & 31;
  const int wid  = (blockIdx.x * blockDim.x + threadIdx.x) >> 5;
  if (wid >= N_TOK * NH) return;
  const int n = wid >> 3, h = wid & 7, d = lane;
  const size_t rb = (size_t)n * (3 * CDIM);
  const size_t ob = (size_t)n * CDIM + h * HD + d;

  float q = qkv[rb + h * HD + d];
  float k = qkv[rb + CDIM + h * HD + d];
  float sq = q * q, sk = k * k;
#pragma unroll
  for (int m = 1; m < 32; m <<= 1) {          // full-wave butterfly (wave32)
    sq += __shfl_xor(sq, m, 32);
    sk += __shfl_xor(sk, m, 32);
  }
  qn[ob] = (__bf16)(q / (sqrtf(sq) + 1e-8f));
  kn[ob] = (__bf16)(k / (sqrtf(sk) + 1e-8f));

  if (is_cls) {
    float v = qkv[rb + 2 * CDIM + h * HD + d];
    float sv = v * v;
#pragma unroll
    for (int m = 1; m < 32; m <<= 1) sv += __shfl_xor(sv, m, 32);
    vn[ob]  = (__bf16)(v / (sqrtf(sv) + 1e-8f));
    vbf[ob] = (__bf16)v;
    out_x[(size_t)n * (2 * CDIM) + CDIM + h * HD + d] = v;   // x_ori half
  }
}

// ------------- Kernel 2: fused dual-softmax attention + attn@v -------------
// grid = 125 row tiles; 8 waves per block; heads looped inside the block so
// attn_sum rows are block-owned (no atomics -> bitwise deterministic).
__global__ __launch_bounds__(256) void k_attn(
    const __bf16* __restrict__ qn_cls, const __bf16* __restrict__ kn_cls,
    const __bf16* __restrict__ qn_reg, const __bf16* __restrict__ kn_reg,
    const __bf16* __restrict__ vbf,
    float* __restrict__ attn_sum, float* __restrict__ out_x) {
  __shared__ float red_c[8][16], red_r[8][16];
  __shared__ float rowsum_c[16], rowsum_r[16];
  __shared__ float a_tile[8][16][32];   // per-wave attn slab (C->A transpose)
  __shared__ float x_part[8][16][32];   // per-wave partial x

  const int lane = threadIdx.x & 31, w = threadIdx.x >> 5;
  const int half = lane >> 4, nl = lane & 15;
  const int r0 = blockIdx.x * 16;

  for (int h = 0; h < NH; ++h) {
    const int hc = h * HD;
    v16bf aqc = loadA_bf16(qn_cls, CDIM, r0, hc, lane);
    v16bf aqr = loadA_bf16(qn_reg, CDIM, r0, hc, lane);

    // ---- pass 1: exp row sums (fixed max = 25, scores <= 25) ----
    float pc[8], pr[8];
#pragma unroll
    for (int i = 0; i < 8; ++i) { pc[i] = 0.f; pr[i] = 0.f; }
    for (int s = w; s < 63; s += 8) {
#pragma unroll
      for (int ct = 0; ct < 2; ++ct) {
        const int cb = s * 32 + ct * 16;
        if (cb >= N_TOK) continue;            // only tile (62,1) is invalid
        v16bf bc = loadA_bf16(kn_cls, CDIM, cb, hc, lane);  // B = k^T
        v8f cc = {}; cc = wmma_bf16(aqc, bc, cc);
        v16bf br = loadA_bf16(kn_reg, CDIM, cb, hc, lane);
        v8f cr = {}; cr = wmma_bf16(aqr, br, cr);
#pragma unroll
        for (int i = 0; i < 8; ++i) {
          pc[i] += __expf(25.f * cc[i] - 25.f);
          pr[i] += __expf(25.f * cr[i] - 25.f);
        }
      }
    }
#pragma unroll
    for (int m = 1; m < 16; m <<= 1) {        // sum over N-lanes (16-group)
#pragma unroll
      for (int i = 0; i < 8; ++i) {
        pc[i] += __shfl_xor(pc[i], m, 32);
        pr[i] += __shfl_xor(pr[i], m, 32);
      }
    }
    if (nl == 0) {
#pragma unroll
      for (int i = 0; i < 8; ++i) {
        red_c[w][half * 8 + i] = pc[i];
        red_r[w][half * 8 + i] = pr[i];
      }
    }
    __syncthreads();
    if (threadIdx.x < 16) {
      float sc = 0.f, sr = 0.f;
#pragma unroll
      for (int ww = 0; ww < 8; ++ww) { sc += red_c[ww][threadIdx.x]; sr += red_r[ww][threadIdx.x]; }
      rowsum_c[threadIdx.x] = sc;
      rowsum_r[threadIdx.x] = sr;
    }
    __syncthreads();

    // ---- pass 2: attn values -> attn_sum, LDS transpose, attn @ v ----
    v8f x0 = {}, x1 = {};
    for (int s = w; s < 63; s += 8) {
      const int sb = s * 32;
#pragma unroll
      for (int ct = 0; ct < 2; ++ct) {
        const int cb = sb + ct * 16;
        if (cb >= N_TOK) {                    // zero-pad tail slab columns
#pragma unroll
          for (int i = 0; i < 8; ++i) a_tile[w][i + 8 * half][ct * 16 + nl] = 0.f;
          continue;
        }
        v16bf bc = loadA_bf16(kn_cls, CDIM, cb, hc, lane);
        v8f cc = {}; cc = wmma_bf16(aqc, bc, cc);
        v16bf br = loadA_bf16(kn_reg, CDIM, cb, hc, lane);
        v8f cr = {}; cr = wmma_bf16(aqr, br, cr);
        const int col = cb + nl;
#pragma unroll
        for (int i = 0; i < 8; ++i) {
          const int ml = i + 8 * half;
          const int row = r0 + ml;
          float a = 0.5f * (__expf(25.f * cc[i] - 25.f) / rowsum_c[ml] +
                            __expf(25.f * cr[i] - 25.f) / rowsum_r[ml]);
          const int bs = (row / 10) * 10;     // block-diag zeroing, keep diag
          const bool keep = !((col >= bs) && (col < bs + 9)) || (row == col);
          a = keep ? a : 0.f;
          const size_t idx = (size_t)row * N_PAD + col;
          attn_sum[idx] = (h == 0) ? a : (attn_sum[idx] + a);  // block-owned
          a_tile[w][ml][ct * 16 + nl] = a;
        }
      }
      // intra-wave LDS RAW: DS ops from one wave execute in order (ISA 7.3)
      v16bf af = loadA_f32(&a_tile[w][0][0], 32, 0, 0, lane);
      v16bf b0 = loadB_bf16_clamp(vbf, CDIM, sb, hc + 0,  lane);
      v16bf b1 = loadB_bf16_clamp(vbf, CDIM, sb, hc + 16, lane);
      x0 = wmma_bf16(af, b0, x0);
      x1 = wmma_bf16(af, b1, x1);
    }
#pragma unroll
    for (int i = 0; i < 8; ++i) {
      x_part[w][i + 8 * half][nl]      = x0[i];
      x_part[w][i + 8 * half][16 + nl] = x1[i];
    }
    __syncthreads();
    for (int e = threadIdx.x; e < 512; e += 256) {   // fixed-order wave reduce
      const int ml = e >> 5, d = e & 31;
      float sum = 0.f;
#pragma unroll
      for (int ww = 0; ww < 8; ++ww) sum += x_part[ww][ml][d];
      out_x[(size_t)(r0 + ml) * (2 * CDIM) + hc + d] = sum;
    }
    __syncthreads();
  }
}

// --------- Kernel 3a: raw_mean = (1/8) Vn @ Vn^T  (K = 8 heads * 32) -------
// Valid because each head's 32-vector is normalized independently, so the
// head-mean of per-head dots equals the concatenated 256-dot / 8.
__global__ __launch_bounds__(256) void k_rawsim(const __bf16* __restrict__ vn,
                                                float* __restrict__ raw) {
  const int lane = threadIdx.x & 31;
  const int wid  = (blockIdx.x * blockDim.x + threadIdx.x) >> 5;
  if (wid >= 125 * 125) return;
  const int r0 = (wid / 125) * 16, c0 = (wid % 125) * 16;
  v8f acc = {};
#pragma unroll
  for (int kk = 0; kk < CDIM; kk += 32) {
    v16bf a = loadA_bf16(vn, CDIM, r0, kk, lane);
    v16bf b = loadA_bf16(vn, CDIM, c0, kk, lane);   // B = Vn^T
    acc = wmma_bf16(a, b, acc);
  }
  const int n = c0 + (lane & 15), half = lane >> 4;
#pragma unroll
  for (int i = 0; i < 8; ++i)
    raw[(size_t)(r0 + i + 8 * half) * N_PAD + n] = acc[i] * 0.125f;
}

// ------------- Kernel 3b: row softmax + sim-mask renormalization -----------
__device__ __forceinline__ float blk_max(float v, float* red) {
  const int t = threadIdx.x;
  red[t] = v; __syncthreads();
  for (int s = 128; s > 0; s >>= 1) {
    if (t < s) red[t] = fmaxf(red[t], red[t + s]);
    __syncthreads();
  }
  float r = red[0]; __syncthreads();
  return r;
}
__device__ __forceinline__ float blk_sum(float v, float* red) {
  const int t = threadIdx.x;
  red[t] = v; __syncthreads();
  for (int s = 128; s > 0; s >>= 1) {
    if (t < s) red[t] += red[t + s];
    __syncthreads();
  }
  float r = red[0]; __syncthreads();
  return r;
}

__global__ __launch_bounds__(256) void k_sim(const float* __restrict__ attn_sum,
                                             const float* __restrict__ raw,
                                             float* __restrict__ out_sim) {
  __shared__ float row[N_TOK];
  __shared__ float red[256];
  const int r = blockIdx.x, t = threadIdx.x;
  const float* src = attn_sum + (size_t)r * N_PAD;
  for (int c = t; c < N_TOK; c += 256) row[c] = src[c] * 0.125f;  // mean heads
  __syncthreads();

  float m = -3.4e38f;
  for (int c = t; c < N_TOK; c += 256) m = fmaxf(m, row[c]);
  m = blk_max(m, red);

  float s = 0.f;
  for (int c = t; c < N_TOK; c += 256) s += __expf(row[c] - m);
  s = blk_sum(s, red);

  const float* rr = raw + (size_t)r * N_PAD;
  float tsum = 0.f;
  for (int c = t; c < N_TOK; c += 256) {
    const float p  = __expf(row[c] - m) / s;
    const float nu = (rr[c] > 0.75f) ? p : 0.f;
    row[c] = nu;
    tsum += nu;
  }
  tsum = blk_sum(tsum, red);

  const float inv = 1.f / (tsum + 1e-8f);
  for (int c = t; c < N_TOK; c += 256)
    out_sim[(size_t)r * N_TOK + c] = row[c] * inv;
}

// ---------------------------------------------------------------------------
extern "C" void kernel_launch(void* const* d_in, const int* in_sizes, int n_in,
                              void* d_out, int out_size, void* d_ws, size_t ws_size,
                              hipStream_t stream) {
  const float* x_cls = (const float*)d_in[0];
  const float* x_reg = (const float*)d_in[1];
  const float* w_cls = (const float*)d_in[2];
  const float* w_reg = (const float*)d_in[3];
  float* out = (float*)d_out;

  // workspace layout (bytes), total ~51.2 MB
  char* ws = (char*)d_ws;
  constexpr size_t SZ_QKV = (size_t)N_TOK * 3 * CDIM * 4;   // 6,144,000
  constexpr size_t SZ_BF  = (size_t)N_TOK * CDIM * 2;       // 1,024,000
  constexpr size_t SZ_NN  = (size_t)N_TOK * N_PAD * 4;      // 16,384,000
  float*  qkv_cls = (float*)(ws);
  float*  qkv_reg = (float*)(ws + SZ_QKV);
  __bf16* qn_cls  = (__bf16*)(ws + 2 * SZ_QKV);
  __bf16* kn_cls  = (__bf16*)(ws + 2 * SZ_QKV + 1 * SZ_BF);
  __bf16* qn_reg  = (__bf16*)(ws + 2 * SZ_QKV + 2 * SZ_BF);
  __bf16* kn_reg  = (__bf16*)(ws + 2 * SZ_QKV + 3 * SZ_BF);
  __bf16* vn      = (__bf16*)(ws + 2 * SZ_QKV + 4 * SZ_BF);
  __bf16* vbf     = (__bf16*)(ws + 2 * SZ_QKV + 5 * SZ_BF);
  float*  attn_s  = (float*)(ws + 2 * SZ_QKV + 6 * SZ_BF);
  float*  raw     = (float*)(ws + 2 * SZ_QKV + 6 * SZ_BF + SZ_NN);

  // 1) QKV GEMMs: 2*125*48 = 12000 tiles, 1 wave each, 8 waves/block
  k_qkv_gemm<<<1500, 256, 0, stream>>>(x_cls, x_reg, w_cls, w_reg, qkv_cls, qkv_reg);
  // 1b) normalize + pack (+ write x_ori): 16000 waves each
  k_norm_pack<<<2000, 256, 0, stream>>>(qkv_cls, 1, qn_cls, kn_cls, vn, vbf, out);
  k_norm_pack<<<2000, 256, 0, stream>>>(qkv_reg, 0, qn_reg, kn_reg, nullptr, nullptr, nullptr);
  // 2) fused attention: one block per 16-row tile
  k_attn<<<125, 256, 0, stream>>>(qn_cls, kn_cls, qn_reg, kn_reg, vbf, attn_s, out);
  // 3a) v.vT similarity mean: 15625 tiles
  k_rawsim<<<(125 * 125 + 7) / 8, 256, 0, stream>>>(vn, raw);
  // 3b) row-wise masked softmax -> sim output
  k_sim<<<N_TOK, 256, 0, stream>>>(attn_s, raw, out + (size_t)N_TOK * 2 * CDIM);
}
